// transformer_71554155151418
// MI455X (gfx1250) — matrix operand order repaired and checked by reading, not compiled, not run on hardware
//
#include <hip/hip_runtime.h>
#include <hip/hip_bf16.h>
#include <math.h>

// ---------------------------------------------------------------------------
// MI455X (gfx1250) transformer encoder inference.
// All large GEMMs run on v_wmma_f32_16x16x32_f16 (f16 in, f32 accum) with
// double-buffered LDS tiles. Tile staging uses CDNA5 async global->LDS copies
// (GLOBAL_LOAD_ASYNC_TO_LDS_B128, ASYNCcnt) when the toolchain exposes the
// builtins, falling back to b128 loads + ds_store otherwise. GEMM modes are
// compile-time template parameters so the inner loop carries no flag checks.
// Attention is fused (scores -> norm -> softmax -> context) per 16-row tile.
// ---------------------------------------------------------------------------

typedef _Float16 f16;
typedef __attribute__((ext_vector_type(16))) _Float16 v16h;
typedef __attribute__((ext_vector_type(8)))  _Float16 v8h;
typedef __attribute__((ext_vector_type(8)))  float    v8f;
typedef int v4i __attribute__((vector_size(16)));   // matches async-LDS builtin

#if defined(__has_builtin)
#  if __has_builtin(__builtin_amdgcn_global_load_async_to_lds_b128) && \
      __has_builtin(__builtin_amdgcn_s_wait_asynccnt)
#    define USE_ASYNC_LDS 1
#  endif
#endif
#ifndef USE_ASYNC_LDS
#  define USE_ASYNC_LDS 0
#endif

#define GAS __attribute__((address_space(1)))   // global
#define LAS __attribute__((address_space(3)))   // LDS

union AFrag { v16h v; f16 e[16]; };

__device__ inline v8f wmma16(const AFrag& a, const AFrag& b, v8f c) {
  // 8 args: (neg_a, A, neg_b, B, c_mod, C, reuse_a, reuse_b)
  return __builtin_amdgcn_wmma_f32_16x16x32_f16(false, a.v, false, b.v,
                                                (short)0, c, false, false);
}

__device__ inline float wred_sum(float x) {
#pragma unroll
  for (int m = 16; m >= 1; m >>= 1) x += __shfl_xor(x, m, 32);
  return x;
}
__device__ inline float wred_max(float x) {
#pragma unroll
  for (int m = 16; m >= 1; m >>= 1) x = fmaxf(x, __shfl_xor(x, m, 32));
  return x;
}

// ---------------------------------------------------------------------------
// WMMA GEMM: C[M,N] = A[M,K](f16) * B (+bias) (+relu)
//   TRANSB==0: B[k][n] = Bw[k*N + n]   (row-major [K,N])
//   TRANSB==1: B[k][n] = Bw[n*K + k]   (row-major [N,K], i.e. x @ W^T)
// Block: 256 thr = 8 waves (4 m-waves x 2 n-waves). Block tile 128x64,
// wave tile 32x32 (4 accumulators). K-step 32. Double-buffered LDS tiles:
// iteration i stages tile i+1 (async DMA to LDS, or regs+ds_store fallback)
// while the 4 WMMAs consume buffer i&1. One barrier per iteration; the
// per-wave s_wait_asynccnt before the barrier makes all stages visible.
// LDS row strides (40 / 72 f16) keep v8h stores 16B aligned and skew banks.
// Fragment layouts per CDNA5 ISA 7.12.2 (wave32).
// QKV==1 scatters the f16 store to [head=n%8][row][hd=n/8].
// ---------------------------------------------------------------------------
template<int TRANSB, int HAS_BIAS, int RELU, int OUTF, int OUTH, int QKV>
__global__ __launch_bounds__(256)
void gemm_kernel(const f16* __restrict__ A, const f16* __restrict__ Bw,
                 int M, int N, int K,
                 const float* __restrict__ bias,
                 float* __restrict__ outF, f16* __restrict__ outH)
{
  __shared__ f16 As[2][128 * 40];         // [m][k], stride 40  (2 x 10 KB)
  __shared__ f16 Bs[2][2560];             // [k][n] s=72 or [n][k] s=40 (2 x 5 KB)

  int tid = threadIdx.x;
  int lane = tid & 31, wave = tid >> 5;
  int wm = wave & 3, wn = wave >> 2;      // 4 x 2 wave grid
  int m0 = blockIdx.y * 128, n0 = blockIdx.x * 64;
  int lm = lane & 15, hi = lane >> 4;

  // Tile loader coordinates (all 16B-aligned b128 accesses).
  int ar0 = tid >> 2, ac = (tid & 3) * 8;          // A: groups of 8 f16 along K
  int btr = tid >> 2, btc = (tid & 3) * 8;         // B trans: [n][k]
  int bnr = tid >> 3, bnc = (tid & 7) * 8;         // B normal: [k][n]

  const f16* aptr0 = A + (size_t)(m0 + ar0) * K + ac;
  const f16* aptr1 = A + (size_t)(m0 + ar0 + 64) * K + ac;
  const f16* bptr  = TRANSB ? (Bw + (size_t)(n0 + btr) * K + btc)
                            : (Bw + (size_t)bnr * N + n0 + bnc);

  v8f acc[2][2] = {};
  int nk = K >> 5;

  // Fragment construction + 4 WMMAs from buffer `buf`.
  auto compute = [&](int buf) {
    const f16* as = As[buf];
    const f16* bs = Bs[buf];
    AFrag a0, a1, b0, b1;
#pragma unroll
    for (int v = 0; v < 8; ++v) {
      int kb = ((v & 4) ? 16 : 0) + hi * 8 + (v & 3) * 2;
      int mi0 = (wm * 32 + lm) * 40 + kb;
      int mi1 = (wm * 32 + 16 + lm) * 40 + kb;
      a0.e[2 * v] = as[mi0]; a0.e[2 * v + 1] = as[mi0 + 1];
      a1.e[2 * v] = as[mi1]; a1.e[2 * v + 1] = as[mi1 + 1];
    }
    int nb0 = wn * 32 + lm, nb1 = nb0 + 16;
#pragma unroll
    for (int v = 0; v < 8; ++v) {
      int kb = ((v & 4) ? 16 : 0) + hi * 8 + (v & 3) * 2;
      if (TRANSB) {
        b0.e[2 * v] = bs[nb0 * 40 + kb]; b0.e[2 * v + 1] = bs[nb0 * 40 + kb + 1];
        b1.e[2 * v] = bs[nb1 * 40 + kb]; b1.e[2 * v + 1] = bs[nb1 * 40 + kb + 1];
      } else {
        b0.e[2 * v] = bs[kb * 72 + nb0]; b0.e[2 * v + 1] = bs[(kb + 1) * 72 + nb0];
        b1.e[2 * v] = bs[kb * 72 + nb1]; b1.e[2 * v + 1] = bs[(kb + 1) * 72 + nb1];
      }
    }
    acc[0][0] = wmma16(a0, b0, acc[0][0]);
    acc[0][1] = wmma16(a0, b1, acc[0][1]);
    acc[1][0] = wmma16(a1, b0, acc[1][0]);
    acc[1][1] = wmma16(a1, b1, acc[1][1]);
  };

#if USE_ASYNC_LDS
  // CDNA5 async DMA: global -> LDS without a VGPR round trip (ASYNCcnt).
  auto stage = [&](int k0, int buf) {
    __builtin_amdgcn_global_load_async_to_lds_b128(
        (GAS v4i*)(aptr0 + k0),
        (LAS v4i*)&As[buf][ar0 * 40 + ac], 0, 0);
    __builtin_amdgcn_global_load_async_to_lds_b128(
        (GAS v4i*)(aptr1 + k0),
        (LAS v4i*)&As[buf][(ar0 + 64) * 40 + ac], 0, 0);
    if (TRANSB)
      __builtin_amdgcn_global_load_async_to_lds_b128(
          (GAS v4i*)(bptr + k0),
          (LAS v4i*)&Bs[buf][btr * 40 + btc], 0, 0);
    else
      __builtin_amdgcn_global_load_async_to_lds_b128(
          (GAS v4i*)(bptr + (size_t)k0 * N),
          (LAS v4i*)&Bs[buf][bnr * 72 + bnc], 0, 0);
  };
  stage(0, 0);
  for (int i = 0; i < nk; ++i) {
    __builtin_amdgcn_s_wait_asynccnt(0);   // this wave's stage of tile i done
    __syncthreads();                       // ... and everyone else's
    if (i + 1 < nk) stage((i + 1) << 5, (i + 1) & 1);  // DMA overlaps WMMAs
    compute(i & 1);
  }
#else
  v8h ra0, ra1, rb;
  auto gload = [&](int k0) {
    ra0 = *(const v8h*)(aptr0 + k0);
    ra1 = *(const v8h*)(aptr1 + k0);
    if (TRANSB) rb = *(const v8h*)(bptr + k0);
    else        rb = *(const v8h*)(bptr + (size_t)k0 * N);
  };
  auto lstore = [&](int buf) {
    *(v8h*)&As[buf][ar0 * 40 + ac]        = ra0;
    *(v8h*)&As[buf][(ar0 + 64) * 40 + ac] = ra1;
    if (TRANSB) *(v8h*)&Bs[buf][btr * 40 + btc] = rb;
    else        *(v8h*)&Bs[buf][bnr * 72 + bnc] = rb;
  };
  gload(0);
  lstore(0);
  for (int i = 0; i < nk; ++i) {
    __syncthreads();
    if (i + 1 < nk) gload((i + 1) << 5);   // stage next tile into registers
    compute(i & 1);
    if (i + 1 < nk) lstore((i + 1) & 1);   // hidden behind the WMMAs
  }
#endif

#pragma unroll
  for (int sm = 0; sm < 2; ++sm)
#pragma unroll
    for (int sn = 0; sn < 2; ++sn)
#pragma unroll
      for (int r = 0; r < 8; ++r) {
        int m = m0 + wm * 32 + sm * 16 + hi * 8 + r;
        int n = n0 + wn * 32 + sn * 16 + lm;
        float vv = acc[sm][sn][r];
        if (HAS_BIAS) vv += bias[n];
        if (RELU) vv = fmaxf(vv, 0.f);
        if (OUTF) outF[(size_t)m * N + n] = vv;
        if (OUTH) {
          if (QKV) {  // channel c -> head c%8, hd c/8 ; store [head][row][hd]
            int head = n & 7, hd = n >> 3;
            outH[((size_t)head * (size_t)M + (size_t)m) * 64 + hd] = (f16)vv;
          } else {
            outH[(size_t)m * N + n] = (f16)vv;
          }
        }
      }
}

// ---------------------------------------------------------------------------
// Fused attention for one (h, b, 16-row s-tile), one wave per block:
//   e = (q kT)/sqrt(512) -> LDS[16][512] (WMMA, K=64, K-chunks staged in LDS)
//   per-row layernorm(ddof=1, gamma_a/beta_a) then stable softmax (in LDS)
//   out[16,64] = alpha @ V (WMMA, K=512, V-chunks staged in LDS)
// q/k/v in [H][B*S][64] f16 layout; output f16 head-concat [B*S][512].
// ---------------------------------------------------------------------------
__global__ __launch_bounds__(32)
void attn_kernel(const f16* __restrict__ qb, const f16* __restrict__ kb_,
                 const f16* __restrict__ vb, f16* __restrict__ aout,
                 const float* __restrict__ ga, const float* __restrict__ ba,
                 int layer)
{
  __shared__ float es[16][512];   // 32 KB score/alpha tile
  __shared__ f16 Qs[16 * 72];     // 2.25 KB
  __shared__ f16 Ks[16 * 72];     // 2.25 KB
  __shared__ f16 Vs[32 * 72];     // 4.5 KB
  int lane = threadIdx.x & 31;
  int lm = lane & 15, hi = lane >> 4;
  int hb = blockIdx.y;
  int h = hb >> 4, b = hb & 15;
  int s0 = blockIdx.x * 16;
  size_t base = (size_t)h * 8192 + (size_t)b * 512;

  // ---- stage Q tile 16x64 (b128 coalesced) ----
#pragma unroll
  for (int j = 0; j < 4; ++j) {
    int g = lane + 32 * j;                 // 128 groups of 8 f16
    int row = g >> 3, cg = (g & 7) * 8;
    *(v8h*)&Qs[row * 72 + cg] =
        *(const v8h*)(qb + (base + s0 + row) * 64 + cg);
  }
  __syncthreads();

  AFrag aq[2];
#pragma unroll
  for (int kk = 0; kk < 2; ++kk)
#pragma unroll
    for (int v = 0; v < 8; ++v) {
      int kb = kk * 32 + ((v & 4) ? 16 : 0) + hi * 8 + (v & 3) * 2;
      aq[kk].e[2 * v] = Qs[lm * 72 + kb];
      aq[kk].e[2 * v + 1] = Qs[lm * 72 + kb + 1];
    }

  const float scale = 0.04419417382415922f;  // 1/sqrt(512) (d = hidden dim!)
  for (int t0 = 0; t0 < 512; t0 += 16) {
    __syncthreads();
    // stage K chunk 16x64
#pragma unroll
    for (int j = 0; j < 4; ++j) {
      int g = lane + 32 * j;
      int row = g >> 3, cg = (g & 7) * 8;
      *(v8h*)&Ks[row * 72 + cg] =
          *(const v8h*)(kb_ + (base + t0 + row) * 64 + cg);
    }
    __syncthreads();
    v8f acc = {};
#pragma unroll
    for (int kk = 0; kk < 2; ++kk) {
      AFrag bf;  // B[k=hd][n=t] = K[t][hd]: pairs along hd are contiguous
#pragma unroll
      for (int v = 0; v < 8; ++v) {
        int kb = kk * 32 + ((v & 4) ? 16 : 0) + hi * 8 + (v & 3) * 2;
        bf.e[2 * v] = Ks[lm * 72 + kb];
        bf.e[2 * v + 1] = Ks[lm * 72 + kb + 1];
      }
      acc = wmma16(aq[kk], bf, acc);
    }
#pragma unroll
    for (int r = 0; r < 8; ++r) es[hi * 8 + r][t0 + lm] = acc[r] * scale;
  }
  __syncthreads();

  // ---- per-row norm (ddof=1) + softmax over 512, in LDS ----
  float g = ga[layer * 8 + h], bet = ba[layer * 8 + h];
  for (int r = 0; r < 16; ++r) {
    float s1 = 0.f, s2 = 0.f;
    for (int c = lane; c < 512; c += 32) { float x = es[r][c]; s1 += x; s2 += x * x; }
    s1 = wred_sum(s1); s2 = wred_sum(s2);
    float mu = s1 * (1.f / 512.f);
    float var = (s2 - 512.f * mu * mu) * (1.f / 511.f);
    float inv = rsqrtf(var + 1e-8f);
    float mx = -3.4e38f;
    for (int c = lane; c < 512; c += 32) {
      float x = g * ((es[r][c] - mu) * inv) + bet;
      es[r][c] = x; mx = fmaxf(mx, x);
    }
    mx = wred_max(mx);
    float se = 0.f;
    for (int c = lane; c < 512; c += 32) {
      float ex = expf(es[r][c] - mx); es[r][c] = ex; se += ex;
    }
    se = wred_sum(se);
    float rs = 1.f / se;
    for (int c = lane; c < 512; c += 32) es[r][c] *= rs;
  }
  __syncthreads();

  // ---- out[16,64] = alpha[16,512] @ V[512,64] ----
  v8f o0 = {}, o1 = {}, o2 = {}, o3 = {};
  for (int k0 = 0; k0 < 512; k0 += 32) {
    __syncthreads();
    // stage V chunk 32x64
#pragma unroll
    for (int j = 0; j < 8; ++j) {
      int gr = lane + 32 * j;                // 256 groups of 8 f16
      int row = gr >> 3, cg = (gr & 7) * 8;
      *(v8h*)&Vs[row * 72 + cg] =
          *(const v8h*)(vb + (base + k0 + row) * 64 + cg);
    }
    __syncthreads();
    AFrag af;
#pragma unroll
    for (int v = 0; v < 8; ++v) {
      int kb = k0 + ((v & 4) ? 16 : 0) + hi * 8 + (v & 3) * 2;
      af.e[2 * v] = (f16)es[lm][kb]; af.e[2 * v + 1] = (f16)es[lm][kb + 1];
    }
#pragma unroll
    for (int nt = 0; nt < 4; ++nt) {
      AFrag bf;  // B[k=t][n=hd], natural layout in Vs
#pragma unroll
      for (int v = 0; v < 8; ++v) {
        int kb = ((v & 4) ? 16 : 0) + hi * 8 + (v & 3) * 2;
        bf.e[2 * v] = Vs[kb * 72 + nt * 16 + lm];
        bf.e[2 * v + 1] = Vs[(kb + 1) * 72 + nt * 16 + lm];
      }
      v8f& oo = (nt == 0) ? o0 : (nt == 1) ? o1 : (nt == 2) ? o2 : o3;
      oo = wmma16(af, bf, oo);
    }
  }
#pragma unroll
  for (int nt = 0; nt < 4; ++nt) {
    v8f oo = (nt == 0) ? o0 : (nt == 1) ? o1 : (nt == 2) ? o2 : o3;
#pragma unroll
    for (int r = 0; r < 8; ++r) {
      int srow = s0 + hi * 8 + r;  // concat: channel = h*64 + hd
      aout[((size_t)b * 512 + srow) * 512 + h * 64 + nt * 16 + lm] = (f16)oo[r];
    }
  }
}

// Residual add + layernorm over 512 channels (ddof=1, scalar gamma/beta).
__global__ __launch_bounds__(256)
void addnorm_kernel(const float* __restrict__ A, const float* __restrict__ R,
                    const float* __restrict__ gamma, const float* __restrict__ beta,
                    int gidx, float* __restrict__ outF, f16* __restrict__ outH)
{
  int lane = threadIdx.x & 31, wave = threadIdx.x >> 5;
  int row = blockIdx.x * 8 + wave;
  const float* a = A + (size_t)row * 512;
  const float* r = R + (size_t)row * 512;
  float s1 = 0.f, s2 = 0.f;
  for (int c = lane; c < 512; c += 32) { float x = a[c] + r[c]; s1 += x; s2 += x * x; }
  s1 = wred_sum(s1); s2 = wred_sum(s2);
  float mu = s1 * (1.f / 512.f);
  float var = (s2 - 512.f * mu * mu) * (1.f / 511.f);
  float inv = rsqrtf(var + 1e-8f);
  float g = gamma[gidx], bt = beta[gidx];
  for (int c = lane; c < 512; c += 32) {
    float y = g * ((a[c] + r[c] - mu) * inv) + bt;
    outF[(size_t)row * 512 + c] = y;
    if (outH) outH[(size_t)row * 512 + c] = (f16)y;
  }
}

// Embedding + sinusoidal positions; carry = x1 + x2 = 2*E[tok] + pos.
__global__ __launch_bounds__(256)
void embed_kernel(const int* __restrict__ tok, const float* __restrict__ E,
                  float* __restrict__ carry)
{
  size_t i = (size_t)blockIdx.x * 256 + threadIdx.x;
  if (i >= (size_t)16 * 512 * 512) return;
  int d = (int)(i & 511);
  int row = (int)(i >> 9);
  int s = row & 511;
  float x1 = E[(size_t)tok[row] * 512 + d];
  float expo = (float)(d & ~1) * (1.f / 512.f);
  float angle = (float)s * powf(10000.f, -expo);
  float pos = (d & 1) ? cosf(angle) : sinf(angle);
  carry[i] = 2.f * x1 + pos;
}

__global__ __launch_bounds__(256)
void cvt_kernel(const float* __restrict__ src, f16* __restrict__ dst, size_t n)
{
  size_t i = (size_t)blockIdx.x * 256 + threadIdx.x;
  if (i < n) dst[i] = (f16)src[i];
}

__global__ __launch_bounds__(256)
void accum_kernel(float* __restrict__ carry, const float* __restrict__ outv, size_t n)
{
  size_t i = (size_t)blockIdx.x * 256 + threadIdx.x;
  if (i < n) carry[i] += outv[i];
}

// r[row] = dot(x8[row,:], V_e)
__global__ __launch_bounds__(256)
void ve_kernel(const float* __restrict__ x8, const float* __restrict__ Ve,
               float* __restrict__ r)
{
  int lane = threadIdx.x & 31, wave = threadIdx.x >> 5;
  int row = blockIdx.x * 8 + wave;
  float s = 0.f;
  for (int c = lane; c < 512; c += 32) s += x8[(size_t)row * 512 + c] * Ve[c];
  s = wred_sum(s);
  if (lane == 0) r[row] = s;
}

// z[b,c] = sum_s r[b,s] * Vd[s,c]
__global__ __launch_bounds__(256)
void z_kernel(const float* __restrict__ r, const float* __restrict__ Vd,
              float* __restrict__ z)
{
  int c = blockIdx.x * 256 + threadIdx.x;
  int b = blockIdx.y;
  if (c >= 10000) return;
  float s = 0.f;
  for (int t = 0; t < 512; ++t) s += r[b * 512 + t] * Vd[(size_t)t * 10000 + c];
  z[(size_t)b * 10000 + c] = s;
}

__device__ inline float bred_sum(float v, float* tmp) {
  int lane = threadIdx.x & 31, wave = threadIdx.x >> 5;
  v = wred_sum(v);
  if (lane == 0) tmp[wave] = v;
  __syncthreads();
  float r = (threadIdx.x < 8) ? tmp[threadIdx.x] : 0.f;
  r = wred_sum(r);
  if (threadIdx.x == 0) tmp[0] = r;
  __syncthreads();
  r = tmp[0];
  __syncthreads();
  return r;
}
__device__ inline float bred_max(float v, float* tmp) {
  int lane = threadIdx.x & 31, wave = threadIdx.x >> 5;
  v = wred_max(v);
  if (lane == 0) tmp[wave] = v;
  __syncthreads();
  float r = (threadIdx.x < 8) ? tmp[threadIdx.x] : -3.4e38f;
  r = wred_max(r);
  if (threadIdx.x == 0) tmp[0] = r;
  __syncthreads();
  r = tmp[0];
  __syncthreads();
  return r;
}

// Final layernorm (over 10000, ddof=1) + softmax, one block per batch row.
__global__ __launch_bounds__(256)
void final_kernel(const float* __restrict__ z, const float* __restrict__ gf,
                  const float* __restrict__ bf, float* __restrict__ out)
{
  __shared__ float tmp[8];
  int b = blockIdx.x;
  const float* zr = z + (size_t)b * 10000;
  float s1 = 0.f, s2 = 0.f;
  for (int c = threadIdx.x; c < 10000; c += 256) { float x = zr[c]; s1 += x; s2 += x * x; }
  s1 = bred_sum(s1, tmp); s2 = bred_sum(s2, tmp);
  float mu = s1 * (1.f / 10000.f);
  float var = (s2 - 10000.f * mu * mu) * (1.f / 9999.f);
  float inv = rsqrtf(var + 1e-8f);
  float g = gf[0], bt = bf[0];
  float mx = -3.4e38f;
  for (int c = threadIdx.x; c < 10000; c += 256)
    mx = fmaxf(mx, g * ((zr[c] - mu) * inv) + bt);
  mx = bred_max(mx, tmp);
  float se = 0.f;
  for (int c = threadIdx.x; c < 10000; c += 256)
    se += expf(g * ((zr[c] - mu) * inv) + bt - mx);
  se = bred_sum(se, tmp);
  float rs = 1.f / se;
  for (int c = threadIdx.x; c < 10000; c += 256)
    out[(size_t)b * 10000 + c] = expf(g * ((zr[c] - mu) * inv) + bt - mx) * rs;
}

// ---------------------------------------------------------------------------
extern "C" void kernel_launch(void* const* d_in, const int* in_sizes, int n_in,
                              void* d_out, int out_size, void* d_ws, size_t ws_size,
                              hipStream_t stream)
{
  (void)in_sizes; (void)n_in; (void)out_size; (void)ws_size;
  const int*   tok = (const int*)  d_in[0];
  const float* Emb = (const float*)d_in[1];
  const float* WQ  = (const float*)d_in[2];
  const float* WK  = (const float*)d_in[3];
  const float* WV  = (const float*)d_in[4];
  const float* WO  = (const float*)d_in[5];
  const float* W1  = (const float*)d_in[6];
  const float* B1  = (const float*)d_in[7];
  const float* W2  = (const float*)d_in[8];
  const float* B2  = (const float*)d_in[9];
  const float* GA  = (const float*)d_in[10];
  const float* BA  = (const float*)d_in[11];
  const float* G1  = (const float*)d_in[12];
  const float* Bn1 = (const float*)d_in[13];
  const float* G2  = (const float*)d_in[14];
  const float* Bn2 = (const float*)d_in[15];
  const float* Ve  = (const float*)d_in[16];
  const float* Vd  = (const float*)d_in[17];
  const float* Gf  = (const float*)d_in[18];
  const float* Bf  = (const float*)d_in[19];
  float* out = (float*)d_out;

  const size_t NR = 8192;  // B*S rows

  char* wp = (char*)d_ws;
  auto carve = [&](size_t bytes) -> char* {
    char* p = wp; wp += (bytes + 255) & ~(size_t)255; return p;
  };
  f16*   wq16  = (f16*)  carve(6ull * 512 * 512 * 2);
  f16*   wk16  = (f16*)  carve(6ull * 512 * 512 * 2);
  f16*   wv16  = (f16*)  carve(6ull * 512 * 512 * 2);
  f16*   wo16  = (f16*)  carve(6ull * 512 * 512 * 2);
  f16*   w116  = (f16*)  carve(6ull * 512 * 2048 * 2);
  f16*   w216  = (f16*)  carve(6ull * 2048 * 512 * 2);
  float* carry = (float*)carve(NR * 512 * 4);
  f16*   xin16 = (f16*)  carve(NR * 512 * 2);
  f16*   q16   = (f16*)  carve(NR * 512 * 2);
  f16*   k16   = (f16*)  carve(NR * 512 * 2);
  f16*   v16   = (f16*)  carve(NR * 512 * 2);
  f16*   a16   = (f16*)  carve(NR * 512 * 2);
  float* attnF = (float*)carve(NR * 512 * 4);   // reused as ff2 buffer
  float* x2F   = (float*)carve(NR * 512 * 4);
  f16*   x216  = (f16*)  carve(NR * 512 * 2);
  f16*   ff116 = (f16*)  carve(NR * 2048 * 2);
  float* outF  = (float*)carve(NR * 512 * 4);
  float* rbuf  = (float*)carve(NR * 4);
  float* zbuf  = (float*)carve(16ull * 10000 * 4);

  auto cvt = [&](const float* s, f16* d, size_t n) {
    cvt_kernel<<<dim3((unsigned)((n + 255) / 256)), 256, 0, stream>>>(s, d, n);
  };

  // Weights -> f16 (deterministic per call; no caching).
  cvt(WQ, wq16, 6ull * 512 * 512);
  cvt(WK, wk16, 6ull * 512 * 512);
  cvt(WV, wv16, 6ull * 512 * 512);
  cvt(WO, wo16, 6ull * 512 * 512);
  cvt(W1, w116, 6ull * 512 * 2048);
  cvt(W2, w216, 6ull * 2048 * 512);

  // carry = x1 + x2 (embedding + positional)
  embed_kernel<<<dim3((unsigned)((NR * 512 + 255) / 256)), 256, 0, stream>>>(tok, Emb, carry);

  for (int l = 0; l < 6; ++l) {
    size_t wo  = (size_t)l * 512 * 512;
    size_t w1o = (size_t)l * 512 * 2048;

    cvt(carry, xin16, NR * 512);  // encoder input in f16

    // QKV projections (x @ W^T), f16 scattered to [head][row][hd]
    gemm_kernel<1,0,0,0,1,1><<<dim3(8, 64), 256, 0, stream>>>(
        xin16, wq16 + wo, 8192, 512, 512, nullptr, nullptr, q16);
    gemm_kernel<1,0,0,0,1,1><<<dim3(8, 64), 256, 0, stream>>>(
        xin16, wk16 + wo, 8192, 512, 512, nullptr, nullptr, k16);
    gemm_kernel<1,0,0,0,1,1><<<dim3(8, 64), 256, 0, stream>>>(
        xin16, wv16 + wo, 8192, 512, 512, nullptr, nullptr, v16);

    // Fused attention: scores -> layernorm -> softmax -> context -> concat f16
    attn_kernel<<<dim3(32, 128), 32, 0, stream>>>(q16, k16, v16, a16, GA, BA, l);

    // attn = a @ WO
    gemm_kernel<0,0,0,1,0,0><<<dim3(8, 64), 256, 0, stream>>>(
        a16, wo16 + wo, 8192, 512, 512, nullptr, attnF, nullptr);
    // x2 = LN(attn + x)
    addnorm_kernel<<<1024, 256, 0, stream>>>(attnF, carry, G1, Bn1, l, x2F, x216);
    // ff1 = relu(x2 @ W1 + b1)  (f16 out)
    gemm_kernel<0,1,1,0,1,0><<<dim3(32, 64), 256, 0, stream>>>(
        x216, w116 + w1o, 8192, 2048, 512, B1 + (size_t)l * 2048, nullptr, ff116);
    // ff2 = ff1 @ W2 + b2 (reuse attnF)
    gemm_kernel<0,1,0,1,0,0><<<dim3(8, 64), 256, 0, stream>>>(
        ff116, w216 + w1o, 8192, 512, 2048, B2 + (size_t)l * 512, attnF, nullptr);
    // out = LN(ff2 + x2); dense residual carry += out
    addnorm_kernel<<<1024, 256, 0, stream>>>(attnF, x2F, G2, Bn2, l, outF, nullptr);
    accum_kernel<<<16384, 256, 0, stream>>>(carry, outF, NR * 512);
  }

  // Classifier head: x9 = x8 @ V_e ; z = x9^T @ V_d ; LN + softmax
  ve_kernel<<<1024, 256, 0, stream>>>(outF, Ve, rbuf);
  z_kernel<<<dim3(40, 16), 256, 0, stream>>>(rbuf, Vd, zbuf);
  final_kernel<<<16, 256, 0, stream>>>(zbuf, Gf, Bf, out);
}
